// BioDecoder_18769007084138
// MI455X (gfx1250) — matrix operand order repaired and checked by reading, not compiled, not run on hardware
//
#include <hip/hip_runtime.h>
#include <hip/hip_bf16.h>

// ---------------------------------------------------------------------------
// BioDecoder for MI455X (gfx1250): bf16 WMMA GEMMs + fused epilogues.
// Logits / visual GEMMs use 32x64-per-wave tiling (8 WMMA / K-step) for
// ~3x better compute-per-byte; small per-step GEMMs use 1 tile/wave for
// occupancy.
// ---------------------------------------------------------------------------

typedef __attribute__((ext_vector_type(16))) __bf16 v16bf;
typedef __attribute__((ext_vector_type(8)))  float  v8f;

#define BSZ   128
#define HID   512
#define EMB   128
#define DVIS  256
#define NVIS  196
#define VOC   32000
#define MAXL  32

// ------------------------------ WMMA fragment loaders ----------------------
// A-matrix 16x32 bf16 (ISA 7.12.2): lane m=l%16; VGPR0..3 hold K=c..c+7,
// VGPR4..7 hold K=c+16..c+23, c = (l/16)*8.
__device__ __forceinline__ v16bf load_a_frag(const __bf16* A, int row_stride,
                                             int m_base, int k_base, int lane) {
    const __bf16* p = A + (size_t)(m_base + (lane & 15)) * row_stride
                        + k_base + ((lane >> 4) << 3);
    union { uint4 u[2]; v16bf v; } t;
    t.u[0] = *(const uint4*)(p);        // K = c .. c+7
    t.u[1] = *(const uint4*)(p + 16);   // K = c+16 .. c+23
    return t.v;
}

// B-matrix 32x16 bf16: lane n=l%16; V0..3 hold K=c..c+7, V4..7 hold
// K=c+16..c+23, c = (l/16)*8 (weights stored [N, K] row-major).
__device__ __forceinline__ v16bf load_b_frag(const __bf16* W, int row_stride,
                                             int n_base, int k_base, int lane) {
    const __bf16* p = W + (size_t)(n_base + (lane & 15)) * row_stride
                        + k_base + ((lane >> 4) << 3);
    union { uint4 u[2]; v16bf v; } t;
    t.u[0] = *(const uint4*)(p);
    t.u[1] = *(const uint4*)(p + 16);
    return t.v;
}

#define WMMA_BF16(a, b, c) \
    __builtin_amdgcn_wmma_f32_16x16x32_bf16(false, (a), false, (b), (short)0, (c), false, false)

// ------------------------------ block reductions ---------------------------
__device__ __forceinline__ float block_sum(float v, float* sm) {
    int tid = threadIdx.x;
    __syncthreads();
    sm[tid] = v; __syncthreads();
    for (int s = 128; s > 0; s >>= 1) {
        if (tid < s) sm[tid] += sm[tid + s];
        __syncthreads();
    }
    return sm[0];
}
__device__ __forceinline__ float block_max(float v, float* sm) {
    int tid = threadIdx.x;
    __syncthreads();
    sm[tid] = v; __syncthreads();
    for (int s = 128; s > 0; s >>= 1) {
        if (tid < s) sm[tid] = fmaxf(sm[tid], sm[tid + s]);
        __syncthreads();
    }
    return sm[0];
}

__device__ __forceinline__ float sigm(float x) { return 1.0f / (1.0f + __expf(-x)); }

__device__ __forceinline__ float rng01(unsigned a) {
    a ^= a >> 16; a *= 0x7feb352du;
    a ^= a >> 15; a *= 0x846ca68bu;
    a ^= a >> 16;
    return ((float)(a >> 8) + 0.5f) * (1.0f / 16777216.0f);
}

// ------------------------------ prep kernels -------------------------------
__global__ void cvt_f32_bf16(const float* __restrict__ in, __bf16* __restrict__ out, int n) {
    int i = blockIdx.x * 256 + threadIdx.x;
    if (i < n) out[i] = (__bf16)in[i];
}

__global__ void add_bias_pair(const float* a, const float* b, float* out, int n) {
    int i = blockIdx.x * 256 + threadIdx.x;
    if (i < n) out[i] = a[i] + b[i];
}

__global__ void init_state(const float* __restrict__ thought,
                           float* h0, float* c0, float* h1, float* c1,
                           __bf16* h0bf, __bf16* h1bf, int* tok, int* hist) {
    int i = blockIdx.x * 256 + threadIdx.x;
    if (i < BSZ * HID) {
        float t = thought[i];
        h0[i] = t; c0[i] = 0.0f; h1[i] = t; c1[i] = 0.0f;
        h0bf[i] = (__bf16)t; h1bf[i] = (__bf16)t;
    }
    if (i < BSZ) tok[i] = 1;
    if (i < BSZ * 3) hist[i] = -1;
}

// LayerNorm in-place over rows of 512 (one block / row)
__global__ void ln_rows(float* __restrict__ v, const float* __restrict__ g,
                        const float* __restrict__ b) {
    __shared__ float sm[256];
    float* row = v + (size_t)blockIdx.x * HID;
    int tid = threadIdx.x;
    float x0 = row[tid], x1 = row[tid + 256];
    float mean = block_sum(x0 + x1, sm) * (1.0f / HID);
    float d0 = x0 - mean, d1 = x1 - mean;
    float var = block_sum(d0 * d0 + d1 * d1, sm) * (1.0f / HID);
    float inv = rsqrtf(var + 1e-5f);
    row[tid]       = d0 * inv * g[tid]       + b[tid];
    row[tid + 256] = d1 * inv * g[tid + 256] + b[tid + 256];
}

__global__ void embed_bf16(const float* __restrict__ emb, const int* __restrict__ tok,
                           __bf16* __restrict__ xbf) {
    int i = blockIdx.x * 256 + threadIdx.x;     // BSZ*EMB
    int b = i >> 7, e = i & 127;
    xbf[i] = (__bf16)emb[(size_t)tok[b] * EMB + e];
}

// ------------------------------ 1-tile/wave bf16 WMMA GEMM -----------------
// C[M,N] (f32) = (acc? C:0) + A[M,K]*W[N,K]^T + bias. One wave per 16x16 tile.
// Used for the small per-step GEMMs where wave-parallelism > reuse.
__global__ void gemm_bf16(const __bf16* __restrict__ A, const __bf16* __restrict__ W,
                          float* __restrict__ C, int M, int N, int K,
                          const float* __restrict__ bias, int accumulate) {
    int wave = (blockIdx.x * blockDim.x + threadIdx.x) >> 5;
    int lane = threadIdx.x & 31;
    int ntiles = N >> 4;
    int mt = wave / ntiles;
    int nt = wave - mt * ntiles;
    if (mt * 16 >= M) return;                    // wave-uniform
    v8f acc = {};
    for (int k = 0; k < K; k += 32) {
        v16bf a = load_a_frag(A, K, mt * 16, k, lane);
        v16bf b = load_b_frag(W, K, nt * 16, k, lane);
        acc = WMMA_BF16(a, b, acc);
    }
    int n = nt * 16 + (lane & 15);
    float bs = bias ? bias[n] : 0.0f;
#pragma unroll
    for (int r = 0; r < 8; ++r) {
        int m = mt * 16 + r + ((lane >> 4) << 3);
        size_t idx = (size_t)m * N + n;
        float v = acc[r] + bs;
        if (accumulate) v += C[idx];
        C[idx] = v;
    }
}

// ------------------------------ 32x64-tile/wave bf16 WMMA GEMM -------------
// 2 M-tiles x 4 N-tiles per wave: 8 WMMA per K-step from 6 b128-pair loads.
__global__ void gemm_bf16_big(const __bf16* __restrict__ A, const __bf16* __restrict__ W,
                              float* __restrict__ C, int M, int N, int K,
                              const float* __restrict__ bias) {
    int wave = (blockIdx.x * blockDim.x + threadIdx.x) >> 5;
    int lane = threadIdx.x & 31;
    int ngrp = N >> 6;                           // groups of 4 n-tiles
    int mg = wave / ngrp;
    int ng = wave - mg * ngrp;
    if (mg * 32 >= M) return;                    // wave-uniform
    v8f acc[2][4] = {};
    for (int k = 0; k < K; k += 32) {
        v16bf a0 = load_a_frag(A, K, mg * 32,      k, lane);
        v16bf a1 = load_a_frag(A, K, mg * 32 + 16, k, lane);
#pragma unroll
        for (int j = 0; j < 4; ++j) {
            v16bf b = load_b_frag(W, K, ng * 64 + j * 16, k, lane);
            acc[0][j] = WMMA_BF16(a0, b, acc[0][j]);
            acc[1][j] = WMMA_BF16(a1, b, acc[1][j]);
        }
    }
#pragma unroll
    for (int i = 0; i < 2; ++i) {
#pragma unroll
        for (int j = 0; j < 4; ++j) {
            int n = ng * 64 + j * 16 + (lane & 15);
            float bs = bias ? bias[n] : 0.0f;
#pragma unroll
            for (int r = 0; r < 8; ++r) {
                int m = mg * 32 + i * 16 + r + ((lane >> 4) << 3);
                C[(size_t)m * N + n] = acc[i][j][r] + bs;
            }
        }
    }
}

// ------------------------------ LSTM pointwise update ----------------------
__global__ void lstm_update(const float* __restrict__ z, float* __restrict__ c,
                            float* __restrict__ h, __bf16* __restrict__ hbf) {
    int idx = blockIdx.x * 256 + threadIdx.x;    // BSZ*HID
    int b = idx >> 9, j = idx & 511;
    const float* zb = z + (size_t)b * 4 * HID;
    float i_ = sigm(zb[j]);
    float f_ = sigm(zb[HID + j]);
    float g_ = tanhf(zb[2 * HID + j]);
    float o_ = sigm(zb[3 * HID + j]);
    float c2 = f_ * c[idx] + i_ * g_;
    float h2 = o_ * tanhf(c2);
    c[idx] = c2; h[idx] = h2; hbf[idx] = (__bf16)h2;
}

// ------------------------------ gate + visual attention --------------------
__global__ void gate_attn(const float* __restrict__ tpre, const float* __restrict__ gw2,
                          const float* __restrict__ gb2, const float* __restrict__ h1,
                          const float* __restrict__ vp, __bf16* __restrict__ obf) {
    __shared__ float sm[256];
    __shared__ float o_s[HID];
    __shared__ float sc[NVIS];
    int b = blockIdx.x, tid = threadIdx.x;
    // gate = sigmoid(tanh(tpre) . gw2 + gb2)
    float d = 0.0f;
    for (int j = tid; j < HID; j += 256) d += tanhf(tpre[(size_t)b * HID + j]) * gw2[j];
    float gate = sigm(block_sum(d, sm) + gb2[0]);
    for (int j = tid; j < HID; j += 256) o_s[j] = h1[(size_t)b * HID + j] * gate;
    __syncthreads();
    // attention scores
    const float* vpb = vp + (size_t)b * NVIS * HID;
    for (int n = tid; n < NVIS; n += 256) {
        float s = 0.0f;
        for (int hh = 0; hh < HID; ++hh) s += o_s[hh] * vpb[(size_t)n * HID + hh];
        sc[n] = s * 0.044194173824159216f;       // 1/sqrt(512)
    }
    __syncthreads();
    float mv = (tid < NVIS) ? sc[tid] : -INFINITY;
    mv = block_max(mv, sm);
    float ev = (tid < NVIS) ? __expf(sc[tid] - mv) : 0.0f;
    if (tid < NVIS) sc[tid] = ev;
    float Z = block_sum(ev, sm);
    float invZ = 1.0f / Z;
    __syncthreads();
    for (int hh = tid; hh < HID; hh += 256) {
        float acc = 0.0f;
        for (int n = 0; n < NVIS; ++n) acc += sc[n] * vpb[(size_t)n * HID + hh];
        obf[(size_t)b * HID + hh] = (__bf16)(o_s[hh] + acc * invZ);
    }
}

// ------------------------------ logits GEMM + fused epilogue ---------------
// logits[128,32000] = o @ out_w^T + out_b, repetition penalty, /TEMP.
// 32x64 per wave: 8 WMMA / K-step, weight stream stays in L2 (32.8 MB bf16).
__global__ void logits_wmma(const __bf16* __restrict__ obf, const __bf16* __restrict__ Wbf,
                            const float* __restrict__ out_b, const int* __restrict__ hist,
                            float* __restrict__ logits) {
    int wave = (blockIdx.x * blockDim.x + threadIdx.x) >> 5;
    int lane = threadIdx.x & 31;
    int ng = wave >> 2;                          // 500 groups of 4 vocab tiles
    int mg = wave & 3;                           // 4 groups of 2 batch tiles
    v8f acc[2][4] = {};
    const __bf16* wrow = Wbf + (size_t)(ng * 64 + (lane & 15)) * HID + ((lane >> 4) << 4);
#pragma unroll 4
    for (int k = 0; k < HID; k += 32) {
        // prefetch next K-slab of the (L2-resident) weight tile
        __builtin_prefetch(wrow + k + 32, 0, 1);
        v16bf a0 = load_a_frag(obf, HID, mg * 32,      k, lane);
        v16bf a1 = load_a_frag(obf, HID, mg * 32 + 16, k, lane);
#pragma unroll
        for (int j = 0; j < 4; ++j) {
            v16bf b = load_b_frag(Wbf, HID, ng * 64 + j * 16, k, lane);
            acc[0][j] = WMMA_BF16(a0, b, acc[0][j]);
            acc[1][j] = WMMA_BF16(a1, b, acc[1][j]);
        }
    }
#pragma unroll
    for (int i = 0; i < 2; ++i) {
        int mbase = mg * 32 + i * 16 + ((lane >> 4) << 3);
#pragma unroll
        for (int j = 0; j < 4; ++j) {
            int n = ng * 64 + j * 16 + (lane & 15);
            float bias = out_b[n];
#pragma unroll
            for (int r = 0; r < 8; ++r) {
                int m = mbase + r;
                int t0 = hist[m * 3 + 0], t1 = hist[m * 3 + 1], t2 = hist[m * 3 + 2];
                float v = acc[i][j][r] + bias;
                if (n == t0 || n == t1 || n == t2) v -= 2.0f;
                logits[(size_t)m * VOC + n] = v * 1.25f;   // 1/TEMP, TEMP=0.8
            }
        }
    }
}

// ------------------------------ top-p + Gumbel sampling --------------------
__global__ void sample_topp(const float* __restrict__ logits, int* __restrict__ tok,
                            int* __restrict__ hist, int* __restrict__ out, int step) {
    __shared__ float sm[256];
    __shared__ int   si[256];
    int b = blockIdx.x, tid = threadIdx.x;
    const float* lg = logits + (size_t)b * VOC;
    // softmax stats
    float mx = -INFINITY;
    for (int i = tid; i < VOC; i += 256) mx = fmaxf(mx, lg[i]);
    mx = block_max(mx, sm);
    float s = 0.0f;
    for (int i = tid; i < VOC; i += 256) s += __expf(lg[i] - mx);
    float invZ = 1.0f / block_sum(s, sm);
    // bisection on prob threshold t: keep {p > t} with mass <= TOP_P
    float lo = 0.0f, hi = 1.0f;
    for (int it = 0; it < 22; ++it) {
        float mid = 0.5f * (lo + hi);
        float a = 0.0f;
        for (int i = tid; i < VOC; i += 256) {
            float p = __expf(lg[i] - mx) * invZ;
            a += (p > mid) ? p : 0.0f;
        }
        float S = block_sum(a, sm);
        if (S > 0.9f) lo = mid; else hi = mid;
    }
    float thr = hi;
    // Gumbel-max over the kept nucleus (argmax token always kept)
    float best = -INFINITY; int bi = 0;
    unsigned base = 0x2545F491u ^ (unsigned)(step * 0x9E3779B9u) ^ (unsigned)(b * 0x85EBCA6Bu);
    for (int i = tid; i < VOC; i += 256) {
        float l = lg[i];
        float p = __expf(l - mx) * invZ;
        if (p > thr || l >= mx) {
            float u = rng01(base ^ ((unsigned)i * 0xC2B2AE35u));
            float v = l - __logf(-__logf(u));
            if (v > best) { best = v; bi = i; }
        }
    }
    __syncthreads();
    sm[tid] = best; si[tid] = bi; __syncthreads();
    for (int st = 128; st > 0; st >>= 1) {
        if (tid < st && sm[tid + st] > sm[tid]) { sm[tid] = sm[tid + st]; si[tid] = si[tid + st]; }
        __syncthreads();
    }
    if (tid == 0) {
        int t = si[0];
        hist[b * 3 + 0] = hist[b * 3 + 1];
        hist[b * 3 + 1] = hist[b * 3 + 2];
        hist[b * 3 + 2] = t;
        tok[b] = t;
        out[(size_t)b * MAXL + step] = t;
    }
}

// ---------------------------------------------------------------------------
extern "C" void kernel_launch(void* const* d_in, const int* in_sizes, int n_in,
                              void* d_out, int out_size, void* d_ws, size_t ws_size,
                              hipStream_t stream) {
    const float* thought = (const float*)d_in[0];
    const float* visual  = (const float*)d_in[1];
    const float* emb     = (const float*)d_in[2];
    const float* w_ih0   = (const float*)d_in[3];
    const float* w_hh0   = (const float*)d_in[4];
    const float* b_ih0   = (const float*)d_in[5];
    const float* b_hh0   = (const float*)d_in[6];
    const float* w_ih1   = (const float*)d_in[7];
    const float* w_hh1   = (const float*)d_in[8];
    const float* b_ih1   = (const float*)d_in[9];
    const float* b_hh1   = (const float*)d_in[10];
    const float* gw1     = (const float*)d_in[11];
    const float* gb1     = (const float*)d_in[12];
    const float* gw2     = (const float*)d_in[13];
    const float* gb2     = (const float*)d_in[14];
    const float* out_w   = (const float*)d_in[15];
    const float* out_b   = (const float*)d_in[16];
    const float* vis_w   = (const float*)d_in[17];
    const float* vis_b   = (const float*)d_in[18];
    const float* ln_g    = (const float*)d_in[19];
    const float* ln_b    = (const float*)d_in[20];
    int* out_tok = (int*)d_out;                  // [B, MAXL] int32

    // ---- workspace carve-out -------------------------------------------
    char* p = (char*)d_ws;
    auto carve = [&](size_t bytes) -> void* {
        void* r = (void*)p;
        p += (bytes + 255) & ~(size_t)255;
        return r;
    };
    __bf16* outw_bf = (__bf16*)carve((size_t)VOC * HID * 2);
    __bf16* wih0_bf = (__bf16*)carve((size_t)4 * HID * EMB * 2);
    __bf16* whh0_bf = (__bf16*)carve((size_t)4 * HID * HID * 2);
    __bf16* wih1_bf = (__bf16*)carve((size_t)4 * HID * HID * 2);
    __bf16* whh1_bf = (__bf16*)carve((size_t)4 * HID * HID * 2);
    __bf16* gw1_bf  = (__bf16*)carve((size_t)HID * HID * 2);
    __bf16* visw_bf = (__bf16*)carve((size_t)HID * DVIS * 2);
    __bf16* vf_bf   = (__bf16*)carve((size_t)BSZ * NVIS * DVIS * 2);
    float*  vp      = (float*)carve((size_t)BSZ * NVIS * HID * 4);
    float*  logits  = (float*)carve((size_t)BSZ * VOC * 4);
    float*  z       = (float*)carve((size_t)BSZ * 4 * HID * 4);
    float*  tpre    = (float*)carve((size_t)BSZ * HID * 4);
    float*  h0      = (float*)carve((size_t)BSZ * HID * 4);
    float*  c0      = (float*)carve((size_t)BSZ * HID * 4);
    float*  h1      = (float*)carve((size_t)BSZ * HID * 4);
    float*  c1      = (float*)carve((size_t)BSZ * HID * 4);
    __bf16* h0bf    = (__bf16*)carve((size_t)BSZ * HID * 2);
    __bf16* h1bf    = (__bf16*)carve((size_t)BSZ * HID * 2);
    __bf16* obf     = (__bf16*)carve((size_t)BSZ * HID * 2);
    __bf16* xbf     = (__bf16*)carve((size_t)BSZ * EMB * 2);
    float*  b0c     = (float*)carve((size_t)4 * HID * 4);
    float*  b1c     = (float*)carve((size_t)4 * HID * 4);
    int*    tok     = (int*)carve((size_t)BSZ * 4);
    int*    hist    = (int*)carve((size_t)BSZ * 3 * 4);

    auto cvt = [&](const float* src, __bf16* dst, int n) {
        cvt_f32_bf16<<<(n + 255) / 256, 256, 0, stream>>>(src, dst, n);
    };
    // small GEMM: one 16x16 tile per wave (max parallelism)
    auto gemm = [&](const __bf16* A, const __bf16* W, float* C, int M, int N, int K,
                    const float* bias, int accum) {
        int waves = ((M + 15) / 16) * (N / 16);
        gemm_bf16<<<(waves + 7) / 8, 256, 0, stream>>>(A, W, C, M, N, K, bias, accum);
    };
    // big GEMM: 32x64 tile per wave (compute intensity)
    auto gemm_big = [&](const __bf16* A, const __bf16* W, float* C, int M, int N, int K,
                        const float* bias) {
        int waves = ((M + 31) / 32) * (N / 64);
        gemm_bf16_big<<<(waves + 7) / 8, 256, 0, stream>>>(A, W, C, M, N, K, bias);
    };

    // ---- one-time preprocessing (recomputed every call: deterministic) --
    cvt(out_w, outw_bf, VOC * HID);
    cvt(w_ih0, wih0_bf, 4 * HID * EMB);
    cvt(w_hh0, whh0_bf, 4 * HID * HID);
    cvt(w_ih1, wih1_bf, 4 * HID * HID);
    cvt(w_hh1, whh1_bf, 4 * HID * HID);
    cvt(gw1, gw1_bf, HID * HID);
    cvt(vis_w, visw_bf, HID * DVIS);
    cvt(visual, vf_bf, BSZ * NVIS * DVIS);
    add_bias_pair<<<(4 * HID + 255) / 256, 256, 0, stream>>>(b_ih0, b_hh0, b0c, 4 * HID);
    add_bias_pair<<<(4 * HID + 255) / 256, 256, 0, stream>>>(b_ih1, b_hh1, b1c, 4 * HID);

    // visual projection + LayerNorm: vp = LN(vf @ vis_w^T + vis_b)
    gemm_big(vf_bf, visw_bf, vp, BSZ * NVIS, HID, DVIS, vis_b);
    ln_rows<<<BSZ * NVIS, 256, 0, stream>>>(vp, ln_g, ln_b);

    init_state<<<(BSZ * HID + 255) / 256, 256, 0, stream>>>(thought, h0, c0, h1, c1,
                                                            h0bf, h1bf, tok, hist);

    // ---- autoregressive loop -------------------------------------------
    for (int step = 0; step < MAXL; ++step) {
        embed_bf16<<<(BSZ * EMB + 255) / 256, 256, 0, stream>>>(emb, tok, xbf);
        // LSTM layer 0: z = x@wih0^T + (bih0+bhh0) + h0@whh0^T
        gemm(xbf, wih0_bf, z, BSZ, 4 * HID, EMB, b0c, 0);
        gemm(h0bf, whh0_bf, z, BSZ, 4 * HID, HID, nullptr, 1);
        lstm_update<<<(BSZ * HID + 255) / 256, 256, 0, stream>>>(z, c0, h0, h0bf);
        // LSTM layer 1
        gemm(h0bf, wih1_bf, z, BSZ, 4 * HID, HID, b1c, 0);
        gemm(h1bf, whh1_bf, z, BSZ, 4 * HID, HID, nullptr, 1);
        lstm_update<<<(BSZ * HID + 255) / 256, 256, 0, stream>>>(z, c1, h1, h1bf);
        // gate MLP (first layer) + attention
        gemm(h1bf, gw1_bf, tpre, BSZ, HID, HID, gb1, 0);
        gate_attn<<<BSZ, 256, 0, stream>>>(tpre, gw2, gb2, h1, vp, obf);
        // vocab projection (fused bias / rep-penalty / temperature)
        logits_wmma<<<(VOC / 64) * (BSZ / 32) / 8, 256, 0, stream>>>(obf, outw_bf,
                                                                     out_b, hist, logits);
        // top-p nucleus + Gumbel-max sampling
        sample_topp<<<BSZ, 256, 0, stream>>>(logits, tok, hist, out_tok, step);
    }
    (void)in_sizes; (void)n_in; (void)out_size; (void)ws_size;
}
